// SiLUAttention_18511309046443
// MI455X (gfx1250) — compile-verified, matrix operand order checked by reading, not run on hardware
//
#include <hip/hip_runtime.h>

// ---------------------------------------------------------------------------
// SiLU-attention for MI455X (gfx1250, wave32, WMMA 16x16x32 f16, f32 accum)
// B=2, L=4096, D=512, H=8, d=64
// Attention kernel stages K/V tiles through LDS with the Tensor Data Mover
// (TENSOR_LOAD_TO_LDS, double-buffered, TENSORcnt-synchronized).
// ---------------------------------------------------------------------------

typedef __attribute__((ext_vector_type(16))) _Float16 v16h;
typedef __attribute__((ext_vector_type(8)))  _Float16 v8h;
typedef __attribute__((ext_vector_type(8)))  float    v8f;
typedef __attribute__((ext_vector_type(4)))  unsigned v4u;
typedef __attribute__((ext_vector_type(8)))  int      v8i;
typedef __attribute__((ext_vector_type(4)))  int      v4i;

#define B_     2
#define L_     4096
#define D_     512
#define H_     8
#define HD     64          // head dim
#define BH_    (B_ * H_)   // 16
#define MROWS  (B_ * L_)   // 8192
#define NQKUV  (4 * D_)    // 2048

#if __has_builtin(__builtin_amdgcn_tensor_load_to_lds) && \
    __has_builtin(__builtin_amdgcn_s_wait_tensorcnt)
#define USE_TDM 1
#else
#define USE_TDM 0
#endif

__device__ __forceinline__ float siluf(float x) {
    return x / (1.0f + __expf(-x));
}

// Load one 16-lane-striped WMMA fragment (A layout, or B layout with the
// N-for-M mirror) for f16 16x16x32. Per lane: two contiguous 8-halfword
// chunks at K = half*8 and K = 16 + half*8 of this lane's row/column.
__device__ __forceinline__ v16h load_frag(const _Float16* rowk, int half) {
    v8h c0 = *(const v8h*)(rowk + half * 8);
    v8h c1 = *(const v8h*)(rowk + 16 + half * 8);
    v16h f;
#pragma unroll
    for (int i = 0; i < 8; ++i) { f[i] = c0[i]; f[i + 8] = c1[i]; }
    return f;
}

#if USE_TDM
// Issue one 2-D tensor DMA: global (row-major, `stride` elements between
// rows, 2-byte elements) -> LDS, tile = t0 (contiguous) x t1 rows, packed.
// D# bitfields per CDNA5 ISA 8.3-8.6 (groups 0..3).
__device__ __forceinline__ void tdm_load_2d(unsigned lds_off,
                                            const _Float16* g,
                                            unsigned t0, unsigned t1,
                                            unsigned long stride) {
    unsigned long ga = (unsigned long)(size_t)g;
    v4u g0;
    g0[0] = 1u;                                     // count=1, user D#
    g0[1] = lds_off;                                // lds_addr
    g0[2] = (unsigned)(ga & 0xffffffffu);           // global_addr[31:0]
    g0[3] = (unsigned)((ga >> 32) & 0x01ffffffu)    // global_addr[56:32]
          | (2u << 30);                             // type=2 ("image")
    v8i g1;
    g1[0] = (int)(1u << 16);                        // data_size=1 (2B)
    g1[1] = (int)(t0 << 16);                        // tensor_dim0[15:0]
    g1[2] = (int)(t1 << 16);                        // tensor_dim1[15:0]
    g1[3] = (int)(t0 << 16);                        // tile_dim0 [127:112]
    g1[4] = (int)(t1 & 0xffffu);                    // tile_dim1 [143:128]
    g1[5] = (int)(unsigned)(stride & 0xffffffffu);  // dim0_stride[31:0]
    g1[6] = (int)(unsigned)((stride >> 32) & 0xffffu); // dim0_stride[47:32]
    g1[7] = 0;                                      // dim1_stride = 0 (2D)
    v4i g2;
    g2[0] = 1;                                      // tensor_dim2 = 1
    g2[1] = 1;                                      // tensor_dim3 = 1
    g2[2] = 0;
    g2[3] = 0;                                      // tile_dim3 = 0 (unused)
    v4i g3;
    g3[0] = 0;
    g3[1] = (int)(1u << 16);                        // tensor_dim4 = 1
    g3[2] = 0;
    g3[3] = 0;                                      // tile_dim4 = 0 (unused)
#if defined(__clang_major__) && (__clang_major__ >= 23)
    v8i g4 = {};
    __builtin_amdgcn_tensor_load_to_lds(g0, g1, g2, g3, g4, 0);
#else
    __builtin_amdgcn_tensor_load_to_lds(g0, g1, g2, g3, 0);
#endif
}
#endif

// ---------------------------------------------------------------------------
// Kernel 1: f32 -> f16 conversions + weight transposes (grid-stride)
// ---------------------------------------------------------------------------
__global__ void convert_pack(const float* __restrict__ x,
                             const float* __restrict__ wq,
                             const float* __restrict__ wo,
                             _Float16* __restrict__ xh,
                             _Float16* __restrict__ wqt,
                             _Float16* __restrict__ wot) {
    int tid = blockIdx.x * blockDim.x + threadIdx.x;
    int stride = gridDim.x * blockDim.x;
    for (int i = tid; i < MROWS * D_; i += stride) xh[i] = (_Float16)x[i];
    for (int i = tid; i < NQKUV * D_; i += stride) {
        int n = i / D_, k = i % D_;
        wqt[i] = (_Float16)wq[k * NQKUV + n];
    }
    for (int i = tid; i < D_ * D_; i += stride) {
        int n = i / D_, k = i % D_;
        wot[i] = (_Float16)wo[k * D_ + n];
    }
}

// ---------------------------------------------------------------------------
// Kernel 2: qkuv = silu(x @ W_qkuv), scattered into per-head layouts:
//   qh, kh : f16 [B,H,L,d]   vth : f16 [B,H,d,L]   uf : f32 [B,H,L,d]
// ---------------------------------------------------------------------------
__global__ void gemm_qkuv_silu(const _Float16* __restrict__ xh,
                               const _Float16* __restrict__ wqt,
                               _Float16* __restrict__ qh,
                               _Float16* __restrict__ kh,
                               _Float16* __restrict__ vth,
                               float* __restrict__ uf) {
    const int wave = threadIdx.x >> 5, lane = threadIdx.x & 31;
    const int half = lane >> 4, sub = lane & 15;
    const int rowbase = (blockIdx.x * 8 + wave) * 16;
    const int colbase = blockIdx.y * 64;

    v8f acc[4] = {};
    const _Float16* arow = xh + (long)(rowbase + sub) * D_;

#pragma unroll 4
    for (int k0 = 0; k0 < D_; k0 += 32) {
        v16h a = load_frag(arow + k0, half);
#pragma unroll
        for (int j = 0; j < 4; ++j) {
            v16h b = load_frag(wqt + (long)(colbase + j * 16 + sub) * D_ + k0, half);
            acc[j] = __builtin_amdgcn_wmma_f32_16x16x32_f16(
                false, a, false, b, (short)0, acc[j], false, false);
        }
    }

#pragma unroll
    for (int j = 0; j < 4; ++j) {
#pragma unroll
        for (int r = 0; r < 8; ++r) {
            int row = rowbase + r + 8 * half;
            int col = colbase + j * 16 + sub;
            float v = siluf(acc[j][r]);
            int b = row >> 12, l = row & (L_ - 1);
            int which = col >> 9;                  // 0:q 1:k 2:v 3:u
            int c = col & 511;
            int h = c >> 6, dd = c & 63;
            long bhl = (long)(b * H_ + h) * L_ + l;
            if      (which == 0) qh[bhl * HD + dd] = (_Float16)v;
            else if (which == 1) kh[bhl * HD + dd] = (_Float16)v;
            else if (which == 2) vth[((long)(b * H_ + h) * HD + dd) * L_ + l] = (_Float16)v;
            else                 uf[bhl * HD + dd] = v;
        }
    }
}

// ---------------------------------------------------------------------------
// Kernel 3: flash-style causal SiLU attention + LayerNorm(d) + u-gate.
// grid = (L/128, B*H); block = 256 (8 waves). Wave w owns query rows
// [lb + 16w, lb + 16w + 16). All waves iterate to the block's causal limit;
// K (32x64) and V^T (64x32) chunks are DMAed into LDS by the TDM (wave 0),
// double-buffered and synchronized with s_wait_tensorcnt + barriers.
// ---------------------------------------------------------------------------
__global__ void silu_attention(const _Float16* __restrict__ qh,
                               const _Float16* __restrict__ kh,
                               const _Float16* __restrict__ vth,
                               const float* __restrict__ uf,
                               const float* __restrict__ gamma,
                               const float* __restrict__ beta,
                               _Float16* __restrict__ attn) {
    __shared__ __align__(128) _Float16 kbuf[2][32 * HD];   // 8 KB
    __shared__ __align__(128) _Float16 vbuf[2][HD * 32];   // 8 KB
    __shared__ __align__(16)  _Float16 sbuf[8][16 * 32];   // 8 KB (per wave)

    const int wave = threadIdx.x >> 5, lane = threadIdx.x & 31;
    const int half = lane >> 4, sub = lane & 15;
    const int bh = blockIdx.y;                   // b*8 + h
    const int lb = blockIdx.x * 128;             // block's first query row
    const int l0 = lb + wave * 16;               // wave's strip

    const _Float16* qbase = qh  + (long)bh * L_ * HD;
    const _Float16* kbase = kh  + (long)bh * L_ * HD;
    const _Float16* vtb   = vth + (long)bh * HD * L_;
    _Float16* sw = &sbuf[wave][0];

    // Q fragments for this strip: d = 0..31 and 32..63
    const _Float16* qrow = qbase + (long)(l0 + sub) * HD;
    const v16h qa0 = load_frag(qrow, half);
    const v16h qa1 = load_frag(qrow + 32, half);

    v8f acc[4] = {};
    const int nchunks = lb / 32 + 4;             // covers rows < lb+128

#if USE_TDM
    if (wave == 0) {
        tdm_load_2d((unsigned)(size_t)&kbuf[0][0], kbase, HD, 32, HD);
        tdm_load_2d((unsigned)(size_t)&vbuf[0][0], vtb, 32, HD, L_);
    }
#endif

    for (int c = 0; c < nchunks; ++c) {
        const int m0 = c * 32;
        const int buf = c & 1;

#if USE_TDM
        if (wave == 0) {
            if (c + 1 < nchunks) {
                const int m1 = m0 + 32, nb = buf ^ 1;
                tdm_load_2d((unsigned)(size_t)&kbuf[nb][0],
                            kbase + (long)m1 * HD, HD, 32, HD);
                tdm_load_2d((unsigned)(size_t)&vbuf[nb][0],
                            vtb + m1, 32, HD, L_);
                __builtin_amdgcn_s_wait_tensorcnt(2);  // chunk c complete
            } else {
                __builtin_amdgcn_s_wait_tensorcnt(0);
            }
        }
        __syncthreads();                         // LDS tiles visible to all
#else
        __syncthreads();                         // prior compute done
        {   // cooperative staging: 256 threads x 16B each per tile
            int t = threadIdx.x;
            *(v8h*)&kbuf[buf][t * 8] = *(const v8h*)(kbase + (long)m0 * HD + t * 8);
            int dout = (t * 8) >> 5, mc = (t * 8) & 31;
            *(v8h*)&vbuf[buf][t * 8] = *(const v8h*)(vtb + (long)dout * L_ + m0 + mc);
        }
        __syncthreads();
#endif

        // --- S = silu(QK^T / 8) * causal, two 16-col subtiles -> LDS ------
#pragma unroll
        for (int t = 0; t < 2; ++t) {
            const _Float16* krow = &kbuf[buf][(t * 16 + sub) * HD];
            v16h kb0 = load_frag(krow, half);
            v16h kb1 = load_frag(krow + 32, half);
            v8f s = {};
            s = __builtin_amdgcn_wmma_f32_16x16x32_f16(
                    false, qa0, false, kb0, (short)0, s, false, false);
            s = __builtin_amdgcn_wmma_f32_16x16x32_f16(
                    false, qa1, false, kb1, (short)0, s, false, false);
#pragma unroll
            for (int r = 0; r < 8; ++r) {
                int qi   = l0 + r + 8 * half;            // global query row
                int kcol = m0 + t * 16 + sub;            // global key col
                float v = siluf(s[r] * 0.125f);          // /sqrt(64)
                v = (kcol <= qi) ? v : 0.0f;             // causal mask
                sw[(r + 8 * half) * 32 + t * 16 + sub] = (_Float16)v;
            }
        }
        // --- O += S @ V ---------------------------------------------------
        v16h sa = load_frag(sw + sub * 32, half);
#pragma unroll
        for (int j = 0; j < 4; ++j) {
            v16h vb = load_frag(&vbuf[buf][(j * 16 + sub) * 32], half);
            acc[j] = __builtin_amdgcn_wmma_f32_16x16x32_f16(
                false, sa, false, vb, (short)0, acc[j], false, false);
        }
        __syncthreads();                         // release buffer for DMA
    }

    // --- LayerNorm over d=64 per row, gate by u, store f16 [B,L,D] --------
    const int b = bh >> 3, hh = bh & 7;
#pragma unroll
    for (int r = 0; r < 8; ++r) {
        int qi = l0 + r + 8 * half;
        float s1 = acc[0][r] + acc[1][r] + acc[2][r] + acc[3][r];
        float s2 = acc[0][r] * acc[0][r] + acc[1][r] * acc[1][r] +
                   acc[2][r] * acc[2][r] + acc[3][r] * acc[3][r];
#pragma unroll
        for (int m = 1; m < 16; m <<= 1) {       // stays in 16-lane half
            s1 += __shfl_xor(s1, m, 32);
            s2 += __shfl_xor(s2, m, 32);
        }
        float mu  = s1 * (1.0f / 64.0f);
        float var = s2 * (1.0f / 64.0f) - mu * mu;
        float inv = rsqrtf(var + 1e-5f);
        const float* urow = uf + ((long)bh * L_ + qi) * HD;
        long orow = ((long)b * L_ + qi) * D_ + hh * HD;
#pragma unroll
        for (int j = 0; j < 4; ++j) {
            int dd = j * 16 + sub;
            float o = (acc[j][r] - mu) * inv * gamma[dd] + beta[dd];
            o *= urow[dd];
            attn[orow + dd] = (_Float16)o;
        }
    }
}

// ---------------------------------------------------------------------------
// Kernel 4: out = attn @ W_out   (f16 x f16 -> f32), [8192,512]x[512,512]
// ---------------------------------------------------------------------------
__global__ void gemm_out(const _Float16* __restrict__ attn,
                         const _Float16* __restrict__ wot,
                         float* __restrict__ out) {
    const int wave = threadIdx.x >> 5, lane = threadIdx.x & 31;
    const int half = lane >> 4, sub = lane & 15;
    const int rowbase = (blockIdx.x * 8 + wave) * 16;
    const int colbase = blockIdx.y * 64;

    v8f acc[4] = {};
    const _Float16* arow = attn + (long)(rowbase + sub) * D_;

#pragma unroll 4
    for (int k0 = 0; k0 < D_; k0 += 32) {
        v16h a = load_frag(arow + k0, half);
#pragma unroll
        for (int j = 0; j < 4; ++j) {
            v16h b = load_frag(wot + (long)(colbase + j * 16 + sub) * D_ + k0, half);
            acc[j] = __builtin_amdgcn_wmma_f32_16x16x32_f16(
                false, a, false, b, (short)0, acc[j], false, false);
        }
    }
#pragma unroll
    for (int j = 0; j < 4; ++j)
#pragma unroll
        for (int r = 0; r < 8; ++r)
            out[(long)(rowbase + r + 8 * half) * D_ + colbase + j * 16 + sub] =
                acc[j][r];
}

// ---------------------------------------------------------------------------
extern "C" void kernel_launch(void* const* d_in, const int* in_sizes, int n_in,
                              void* d_out, int out_size, void* d_ws, size_t ws_size,
                              hipStream_t stream) {
    (void)in_sizes; (void)n_in; (void)out_size; (void)ws_size;
    const float* x      = (const float*)d_in[0];
    // d_in[1] = attn_mask (causal tril) — enforced analytically, not read
    const float* W_qkuv = (const float*)d_in[2];
    const float* W_out  = (const float*)d_in[3];
    const float* gamma  = (const float*)d_in[4];
    const float* beta   = (const float*)d_in[5];
    float* out = (float*)d_out;

    size_t off = 0;
    auto carve = [&](size_t bytes) {
        void* p = (char*)d_ws + off;
        off += (bytes + 255) & ~(size_t)255;
        return p;
    };
    _Float16* xh   = (_Float16*)carve((size_t)MROWS * D_ * 2);   // 8 MB
    _Float16* wqt  = (_Float16*)carve((size_t)NQKUV * D_ * 2);   // 2 MB
    _Float16* wot  = (_Float16*)carve((size_t)D_ * D_ * 2);      // 0.5 MB
    _Float16* qh   = (_Float16*)carve((size_t)BH_ * L_ * HD * 2);// 8 MB
    _Float16* kh   = (_Float16*)carve((size_t)BH_ * L_ * HD * 2);// 8 MB
    _Float16* vth  = (_Float16*)carve((size_t)BH_ * HD * L_ * 2);// 8 MB
    float*    uf   = (float*)   carve((size_t)BH_ * L_ * HD * 4);// 16 MB
    _Float16* attn = (_Float16*)carve((size_t)MROWS * D_ * 2);   // 8 MB

    convert_pack<<<512, 256, 0, stream>>>(x, W_qkuv, W_out, xh, wqt, wot);

    gemm_qkuv_silu<<<dim3(MROWS / 128, NQKUV / 64), 256, 0, stream>>>(
        xh, wqt, qh, kh, vth, uf);

    silu_attention<<<dim3(L_ / 128, BH_), 256, 0, stream>>>(
        qh, kh, vth, uf, gamma, beta, attn);

    gemm_out<<<dim3(MROWS / 128, D_ / 64), 256, 0, stream>>>(attn, wot, out);
}
